// BrailleFormer_19542101196884
// MI455X (gfx1250) — matrix operand order repaired and checked by reading, not compiled, not run on hardware
//
#include <hip/hip_runtime.h>
#include <hip/hip_bf16.h>

// ---------------------------------------------------------------------------
// BrailleFormer forward for MI455X (gfx1250, wave32, WMMA bf16)
// ---------------------------------------------------------------------------
// Shapes: B=4, T=3072, V=256, D=768, NH=12, HD=64, L=6, DFF=3072, K=6, R=32
//         N_CELLS=512, grid H=16, W=32, M2 = B*N_CELLS = 2048, BT = 12288
// All GEMMs run on v_wmma_f32_16x16x32_bf16 with f32 accumulation.
// ---------------------------------------------------------------------------

typedef __bf16 v16bf __attribute__((ext_vector_type(16)));
typedef __bf16 v8bf  __attribute__((ext_vector_type(8)));
typedef float  v8f   __attribute__((ext_vector_type(8)));

#define LDSK 40   // padded LDS row stride (bf16 elems): 80B rows, 16B aligned

__device__ __forceinline__ __bf16 f2bf(float f) {
  unsigned u = __builtin_bit_cast(unsigned, f);
  u += 0x7fffu + ((u >> 16) & 1u);               // round-to-nearest-even
  unsigned short h = (unsigned short)(u >> 16);
  return __builtin_bit_cast(__bf16, h);
}

__device__ __forceinline__ v16bf make_v16(v8bf lo, v8bf hi) {
  v16bf r;
#pragma unroll
  for (int i = 0; i < 8; ++i) { r[i] = lo[i]; r[i + 8] = hi[i]; }
  return r;
}

struct GemmArgs {
  const float* A;  const float* B;  float* C;
  const float* A2;    // optional: elementwise multiplier on A (same lda)
  const float* bias;  // optional: [N]
  const float* res;   // optional: residual added before activation
  long lda, ldb, ldc, ldres;
  int  M, N, Kd;
  int  bn1, bn2;      // batch decomposition: z -> (i0, i1, i2)
  long sA0, sA1, sA2, sB0, sB1, sB2, sC0, sC1, sC2, sR0, sR1, sR2;
  float alpha;
  int   act;          // 0 = none, 1 = sigmoid, 2 = silu
};

// Generic tiled GEMM:  C = act(alpha * (A .* A2) @ B + bias + res)
// BLAYOUT 0: B is [K,N] row-major (plain weight tensor, transposed into LDS)
// BLAYOUT 1: B is [N,K] row-major (B^T given: Q@K^T, tied logits head)
template <int BLAYOUT>
__global__ __launch_bounds__(128)
void gemm_wmma(GemmArgs g) {
  __shared__ alignas(16) __bf16 As[128 * LDSK];
  __shared__ alignas(16) __bf16 Bs[128 * LDSK];

  const int tid  = threadIdx.x;
  const int lane = tid & 31;
  const int wave = tid >> 5;

  const int z  = blockIdx.z;
  const int i2 = z % g.bn2;
  const int zt = z / g.bn2;
  const int i1 = zt % g.bn1;
  const int i0 = zt / g.bn1;

  const float* A  = g.A + i0 * g.sA0 + i1 * g.sA1 + i2 * g.sA2;
  const float* B  = g.B + i0 * g.sB0 + i1 * g.sB1 + i2 * g.sB2;
  float*       C  = g.C + i0 * g.sC0 + i1 * g.sC1 + i2 * g.sC2;
  const float* R  = g.res ? g.res + i0 * g.sR0 + i1 * g.sR1 + i2 * g.sR2 : nullptr;
  const float* A2 = g.A2;

  const int m0 = blockIdx.y * 128;
  const int n0 = blockIdx.x * 128;
  const int wm = (wave & 1) * 64;
  const int wn = (wave >> 1) * 64;

  v8f acc[4][4] = {};

  const int lr   = lane & 15;
  const int half = lane >> 4;

  for (int k0 = 0; k0 < g.Kd; k0 += 32) {
    // ---- stage A tile [128 x 32] f32 -> bf16 LDS (thread t = row t) ----
    {
      const int gm = m0 + tid;
      __bf16* dst = &As[tid * LDSK];
      if (gm < g.M) {
        const float* src = A + (long)gm * g.lda + k0;
        if (k0 + 32 < g.Kd) __builtin_prefetch(src + 32, 0, 0);
        if (A2) {
          const float* src2 = A2 + (long)gm * g.lda + k0;
          for (int c = 0; c < 32; ++c) dst[c] = f2bf(src[c] * src2[c]);
        } else {
#pragma unroll
          for (int c = 0; c < 32; c += 4) {
            float4 v = *(const float4*)(src + c);
            dst[c+0] = f2bf(v.x); dst[c+1] = f2bf(v.y);
            dst[c+2] = f2bf(v.z); dst[c+3] = f2bf(v.w);
          }
        }
      } else {
        for (int c = 0; c < 32; ++c) dst[c] = f2bf(0.f);
      }
    }
    // ---- stage B tile -> LDS as Bt[n][k] bf16 ----
    {
      const int gn = n0 + tid;
      __bf16* dst = &Bs[tid * LDSK];
      if (gn < g.N) {
        if (BLAYOUT == 0) {
          const float* src = B + (long)k0 * g.ldb + gn;
          if (k0 + 32 < g.Kd) __builtin_prefetch(src + 32 * g.ldb, 0, 0);
#pragma unroll
          for (int kk = 0; kk < 32; ++kk) dst[kk] = f2bf(src[(long)kk * g.ldb]);
        } else {
          const float* src = B + (long)gn * g.ldb + k0;
          if (k0 + 32 < g.Kd) __builtin_prefetch(src + 32, 0, 0);
#pragma unroll
          for (int kk = 0; kk < 32; kk += 4) {
            float4 v = *(const float4*)(src + kk);
            dst[kk+0] = f2bf(v.x); dst[kk+1] = f2bf(v.y);
            dst[kk+2] = f2bf(v.z); dst[kk+3] = f2bf(v.w);
          }
        }
      } else {
        for (int kk = 0; kk < 32; ++kk) dst[kk] = f2bf(0.f);
      }
    }
    __syncthreads();

    // ---- fragments per CDNA5 WMMA VGPR layouts ----
    // A 16x32 bf16: lanes 0-15 -> row M=lr, K {0..7,16..23}; lanes 16-31 -> K {8..15,24..31}
    v16bf af[4];
#pragma unroll
    for (int mi = 0; mi < 4; ++mi) {
      const __bf16* p = &As[(wm + mi * 16 + lr) * LDSK + half * 8];
      v8bf lo = *(const v8bf*)(p);
      v8bf hi = *(const v8bf*)(p + 16);
      af[mi] = make_v16(lo, hi);
    }
    // B 32x16 bf16: lanes 0-15 -> col N=lr, K {0..7,16..23}; lanes 16-31 -> K {8..15,24..31}
    v16bf bfr[4];
#pragma unroll
    for (int ni = 0; ni < 4; ++ni) {
      const __bf16* p = &Bs[(wn + ni * 16 + lr) * LDSK + half * 8];
      v8bf lo = *(const v8bf*)(p);
      v8bf hi = *(const v8bf*)(p + 16);
      bfr[ni] = make_v16(lo, hi);
    }
#pragma unroll
    for (int mi = 0; mi < 4; ++mi)
#pragma unroll
      for (int ni = 0; ni < 4; ++ni)
        acc[mi][ni] = __builtin_amdgcn_wmma_f32_16x16x32_bf16(
            false, af[mi], false, bfr[ni], (short)0, acc[mi][ni], false, false);
    __syncthreads();
  }

  // ---- epilogue: C 16x16 f32 layout (VGPR r: M = r + 8*half; N = lr) ----
#pragma unroll
  for (int mi = 0; mi < 4; ++mi) {
#pragma unroll
    for (int ni = 0; ni < 4; ++ni) {
      const int col = n0 + wn + ni * 16 + lr;
      if (col >= g.N) continue;
#pragma unroll
      for (int r = 0; r < 8; ++r) {
        const int row = m0 + wm + mi * 16 + half * 8 + r;
        if (row >= g.M) continue;
        float v = acc[mi][ni][r] * g.alpha;
        if (g.bias) v += g.bias[col];
        if (R)      v += R[(long)row * g.ldres + col];
        if (g.act == 1)      v = 1.f / (1.f + __expf(-v));
        else if (g.act == 2) v = v / (1.f + __expf(-v));
        C[(long)row * g.ldc + col] = v;
      }
    }
  }
}

// ---------------------------------------------------------------------------
// Elementwise / normalization / softmax kernels
// ---------------------------------------------------------------------------

__global__ void embed_kernel(const int* __restrict__ ids,
                             const float* __restrict__ tok,
                             const float* __restrict__ pos,
                             float* __restrict__ out, int total) {
  int i = blockIdx.x * 256 + threadIdx.x;
  if (i >= total) return;
  int d = i % 768;
  long row = i / 768;          // b*T + t
  int t = (int)(row % 3072);
  int id = ids[row];
  out[i] = tok[(long)id * 768 + d] + pos[(long)t * 768 + d];
}

__global__ void cellsum_kernel(const float* __restrict__ tokres,
                               float* __restrict__ sum, float* __restrict__ mean,
                               int total) {
  int i = blockIdx.x * 256 + threadIdx.x;
  if (i >= total) return;      // total = 2048*768
  int d = i % 768;
  long row = i / 768;
  const float* p = tokres + row * 6 * 768 + d;
  float s = 0.f;
#pragma unroll
  for (int k = 0; k < 6; ++k) s += p[k * 768];
  sum[i]  = s;
  mean[i] = s * (1.f / 6.f);
}

__global__ void tri_prod_kernel(const float* __restrict__ abc,
                                float* __restrict__ p, int total) {
  int i = blockIdx.x * 256 + threadIdx.x;
  if (i >= total) return;      // total = 2048*32
  p[i] = abc[i] * abc[65536 + i] * abc[131072 + i];
}

__global__ void expand_kernel(const float* __restrict__ cell,
                              const float* __restrict__ dec_pos,
                              float* __restrict__ out, int total) {
  int i = blockIdx.x * 256 + threadIdx.x;
  if (i >= total) return;      // total = 12288*768
  int d = i % 768;
  long row = i / 768;          // b*T + t
  int t = (int)(row % 3072);
  int b = (int)(row / 3072);
  long crow = (long)b * 512 + (t / 6);
  int k = t % 6;
  out[i] = cell[crow * 768 + d] + dec_pos[(long)k * 768 + d];
}

// out = rms(x (+ res), w); 768 cols, one row per block
__global__ __launch_bounds__(256)
void rms_kernel(const float* __restrict__ x, const float* __restrict__ res,
                const float* __restrict__ w, float* __restrict__ out) {
  long row = blockIdx.x;
  const float* xr = x + row * 768;
  const float* rr = res ? res + row * 768 : nullptr;
  float* orow = out + row * 768;
  int tid = threadIdx.x;
  float vals[3]; float local = 0.f;
#pragma unroll
  for (int t = 0; t < 3; ++t) {
    int c = tid + 256 * t;
    float v = xr[c]; if (rr) v += rr[c];
    vals[t] = v; local += v * v;
  }
  __shared__ float red[256];
  red[tid] = local; __syncthreads();
  for (int s = 128; s > 0; s >>= 1) { if (tid < s) red[tid] += red[tid + s]; __syncthreads(); }
  float inv = rsqrtf(red[0] * (1.f / 768.f) + 1e-6f);
#pragma unroll
  for (int t = 0; t < 3; ++t) { int c = tid + 256 * t; orow[c] = vals[t] * inv * w[c]; }
}

// cell = rms(g*tri + (1-g)*resid, w); tri/resid live in encconcat halves
__global__ __launch_bounds__(256)
void enc_combine_rms(const float* __restrict__ gsig, const float* __restrict__ cc,
                     const float* __restrict__ w, float* __restrict__ out) {
  long row = blockIdx.x;
  const float* gr = gsig + row * 768;
  const float* tr = cc + row * 1536;
  const float* rr = tr + 768;
  float* orow = out + row * 768;
  int tid = threadIdx.x;
  float vals[3]; float local = 0.f;
#pragma unroll
  for (int t = 0; t < 3; ++t) {
    int c = tid + 256 * t;
    float gv = gr[c];
    float v = gv * tr[c] + (1.f - gv) * rr[c];
    vals[t] = v; local += v * v;
  }
  __shared__ float red[256];
  red[tid] = local; __syncthreads();
  for (int s = 128; s > 0; s >>= 1) { if (tid < s) red[tid] += red[tid + s]; __syncthreads(); }
  float inv = rsqrtf(red[0] * (1.f / 768.f) + 1e-6f);
#pragma unroll
  for (int t = 0; t < 3; ++t) { int c = tid + 256 * t; orow[c] = vals[t] * inv * w[c]; }
}

// out = rms(rms(x, w2), wf) in one pass:
//   y = x*inv1*w2;  mean(y^2) = inv1^2 * mean((x*w2)^2)  -> both sums in one sweep
__global__ __launch_bounds__(256)
void dec_double_rms(const float* __restrict__ x, const float* __restrict__ w2,
                    const float* __restrict__ wf, float* __restrict__ out) {
  long row = blockIdx.x;
  const float* xr = x + row * 768;
  float* orow = out + row * 768;
  int tid = threadIdx.x;
  float tw[3]; float s1 = 0.f, s2 = 0.f;
#pragma unroll
  for (int t = 0; t < 3; ++t) {
    int c = tid + 256 * t;
    float v = xr[c];
    s1 += v * v;
    float y = v * w2[c];
    tw[t] = y; s2 += y * y;
  }
  __shared__ float redA[256];
  __shared__ float redB[256];
  redA[tid] = s1; redB[tid] = s2; __syncthreads();
  for (int s = 128; s > 0; s >>= 1) {
    if (tid < s) { redA[tid] += redA[tid + s]; redB[tid] += redB[tid + s]; }
    __syncthreads();
  }
  float inv1 = rsqrtf(redA[0] * (1.f / 768.f) + 1e-6f);
  float inv2 = rsqrtf(inv1 * inv1 * redB[0] * (1.f / 768.f) + 1e-6f);
  float sc = inv1 * inv2;
#pragma unroll
  for (int t = 0; t < 3; ++t) { int c = tid + 256 * t; orow[c] = tw[t] * sc * wf[c]; }
}

// Masked softmax over one score row; mask computed analytically from direction.
// grid = (512 rows, 192 batches=(g,b,h)); block = 128, 4 cols/thread.
__global__ __launch_bounds__(128)
void softmax_mask(float* __restrict__ sc) {
  int i = blockIdx.x;          // query row 0..511
  int z = blockIdx.y;          // (g*4 + b)*12 + h
  int gdir = z / 48;
  float* row = sc + ((long)z * 512 + i) * 512;
  int tid = threadIdx.x;
  int cmi = (i & 31) * 16 + (i >> 5);     // column-major index, H=16 W=32
  float v[4]; float mx = -3.4e38f;
#pragma unroll
  for (int r = 0; r < 4; ++r) {
    int j = tid + 128 * r;
    int cmj = (j & 31) * 16 + (j >> 5);
    bool ok = (gdir == 0) ? (j <= i)
            : (gdir == 1) ? (j >= i)
            : (gdir == 2) ? (cmj <= cmi)
                          : (cmj >= cmi);
    v[r] = ok ? row[j] : -3.4e38f;
    mx = fmaxf(mx, v[r]);
  }
  __shared__ float red[128];
  red[tid] = mx; __syncthreads();
  for (int s = 64; s > 0; s >>= 1) { if (tid < s) red[tid] = fmaxf(red[tid], red[tid + s]); __syncthreads(); }
  mx = red[0]; __syncthreads();
  float sum = 0.f;
#pragma unroll
  for (int r = 0; r < 4; ++r) {
    v[r] = (v[r] > -1e37f) ? __expf(v[r] - mx) : 0.f;
    sum += v[r];
  }
  red[tid] = sum; __syncthreads();
  for (int s = 64; s > 0; s >>= 1) { if (tid < s) red[tid] += red[tid + s]; __syncthreads(); }
  float inv = 1.f / red[0];
#pragma unroll
  for (int r = 0; r < 4; ++r) row[tid + 128 * r] = v[r] * inv;
}

// ---------------------------------------------------------------------------
// Host side
// ---------------------------------------------------------------------------

static GemmArgs ga(const float* A, long lda, const float* B, long ldb,
                   float* C, long ldc, int M, int N, int K) {
  GemmArgs g{};
  g.A = A; g.B = B; g.C = C;
  g.lda = lda; g.ldb = ldb; g.ldc = ldc; g.ldres = ldc;
  g.M = M; g.N = N; g.Kd = K;
  g.bn1 = 1; g.bn2 = 1;
  g.alpha = 1.f; g.act = 0;
  return g;
}

template <int BL>
static void launch_gemm(hipStream_t s, const GemmArgs& g, int batch) {
  dim3 grid((g.N + 127) / 128, (g.M + 127) / 128, batch);
  gemm_wmma<BL><<<grid, 128, 0, s>>>(g);
}

extern "C" void kernel_launch(void* const* d_in, const int* in_sizes, int n_in,
                              void* d_out, int out_size, void* d_ws, size_t ws_size,
                              hipStream_t stream) {
  (void)in_sizes; (void)n_in; (void)out_size; (void)ws_size;

  const int*   ids     = (const int*)  d_in[0];
  const float* tok_emb = (const float*)d_in[1];
  const float* pos_emb = (const float*)d_in[2];
  const float* enc_A   = (const float*)d_in[3];
  const float* enc_Bw  = (const float*)d_in[4];
  const float* enc_Cw  = (const float*)d_in[5];
  const float* enc_ow  = (const float*)d_in[6];
  const float* enc_rw  = (const float*)d_in[7];
  const float* enc_gw  = (const float*)d_in[8];
  const float* enc_gb  = (const float*)d_in[9];
  const float* enc_nw  = (const float*)d_in[10];
  const float* ln1_w   = (const float*)d_in[11];
  const float* ln2_w   = (const float*)d_in[12];
  const float* qkv_w   = (const float*)d_in[13];
  const float* aout_w  = (const float*)d_in[14];
  const float* gate_w  = (const float*)d_in[15];
  const float* gate_b  = (const float*)d_in[16];
  const float* fuse_w  = (const float*)d_in[17];
  const float* anrm_w  = (const float*)d_in[18];
  const float* ff1_w   = (const float*)d_in[19];
  const float* ff2_w   = (const float*)d_in[20];
  const float* dec_pos = (const float*)d_in[21];
  const float* dn1_w   = (const float*)d_in[22];
  const float* dlin1   = (const float*)d_in[23];
  const float* dlin2   = (const float*)d_in[24];
  const float* dn2_w   = (const float*)d_in[25];
  const float* lnf_w   = (const float*)d_in[26];
  float* out = (float*)d_out;

  float* ws = (float*)d_ws;
  size_t off = 0;
  auto take = [&](size_t n) { float* p = ws + off; off += n; return p; };

  float* tokres = take((size_t)12288 * 768);   // [B,T,D] token residual
  float* csum   = take((size_t)2048 * 768);
  float* cmean  = take((size_t)2048 * 768);
  float* abc    = take((size_t)3 * 2048 * 32);
  float* pbuf   = take((size_t)2048 * 32);
  float* encc   = take((size_t)2048 * 1536);   // [tri | resid]
  float* gsig   = take((size_t)2048 * 768);
  float* x      = take((size_t)2048 * 768);    // block state
  float* h      = take((size_t)2048 * 768);
  float* qkvb   = take((size_t)4 * 2048 * 2304);
  float* sc     = take((size_t)192 * 512 * 512);
  float* yb     = take((size_t)4 * 2048 * 768);
  float* catb   = take((size_t)2048 * 3072);
  float* gateb  = take((size_t)2048 * 3072);
  float* fusedb = take((size_t)2048 * 768);
  float* x1     = take((size_t)2048 * 768);
  float* hh     = take((size_t)2048 * 768);
  float* ffb    = take((size_t)2048 * 3072);
  float* expb   = take((size_t)12288 * 768);
  float* dech   = take((size_t)12288 * 768);
  float* dt1    = take((size_t)12288 * 768);
  float* dtmp   = take((size_t)12288 * 768);
  float* finalb = take((size_t)12288 * 768);

  // ---- embeddings + token residual ----
  { int total = 12288 * 768;
    embed_kernel<<<(total + 255) / 256, 256, 0, stream>>>(ids, tok_emb, pos_emb, tokres, total); }
  // ---- cell encoder ----
  { int total = 2048 * 768;
    cellsum_kernel<<<(total + 255) / 256, 256, 0, stream>>>(tokres, csum, cmean, total); }
  for (int j = 0; j < 3; ++j) {
    const float* W = (j == 0) ? enc_A : (j == 1) ? enc_Bw : enc_Cw;
    GemmArgs g = ga(csum, 768, W, 32, abc + (size_t)j * 2048 * 32, 32, 2048, 32, 768);
    launch_gemm<0>(stream, g, 1);
  }
  { int total = 2048 * 32;
    tri_prod_kernel<<<(total + 255) / 256, 256, 0, stream>>>(abc, pbuf, total); }
  { GemmArgs g = ga(pbuf, 32, enc_ow, 768, encc, 1536, 2048, 768, 32);
    launch_gemm<0>(stream, g, 1); }                                   // tri
  { GemmArgs g = ga(cmean, 768, enc_rw, 768, encc + 768, 1536, 2048, 768, 768);
    launch_gemm<0>(stream, g, 1); }                                   // resid
  { GemmArgs g = ga(encc, 1536, enc_gw, 768, gsig, 768, 2048, 768, 1536);
    g.bias = enc_gb; g.act = 1; launch_gemm<0>(stream, g, 1); }       // gate
  enc_combine_rms<<<2048, 256, 0, stream>>>(gsig, encc, enc_nw, x);

  // ---- transformer blocks ----
  for (int l = 0; l < 6; ++l) {
    const float* ln1  = ln1_w + (size_t)l * 768;
    const float* ln2  = ln2_w + (size_t)l * 768;
    const float* qkvw = qkv_w + (size_t)l * 4 * 768 * 2304;
    const float* aow  = aout_w + (size_t)l * 4 * 768 * 768;
    const float* gw   = gate_w + (size_t)l * 3072 * 3072;
    const float* gb   = gate_b + (size_t)l * 3072;
    const float* fw   = fuse_w + (size_t)l * 3072 * 768;
    const float* anw  = anrm_w + (size_t)l * 768;
    const float* f1   = ff1_w + (size_t)l * 768 * 3072;
    const float* f2   = ff2_w + (size_t)l * 3072 * 768;

    rms_kernel<<<2048, 256, 0, stream>>>(x, nullptr, ln1, h);
    // qkv: batched over 4 directions (shared A)
    { GemmArgs g = ga(h, 768, qkvw, 2304, qkvb, 2304, 2048, 2304, 768);
      g.sA0 = 0; g.sB0 = (long)768 * 2304; g.sC0 = (long)2048 * 2304;
      launch_gemm<0>(stream, g, 4); }
    // scores = Q @ K^T * scale   (batched over 192 = g*b*h, B pre-transposed)
    { GemmArgs g = ga(qkvb, 2304, qkvb + 768, 2304, sc, 512, 512, 512, 64);
      g.alpha = 0.125f; g.bn1 = 4; g.bn2 = 12;
      g.sA0 = (long)2048 * 2304; g.sA1 = (long)512 * 2304; g.sA2 = 64;
      g.sB0 = (long)2048 * 2304; g.sB1 = (long)512 * 2304; g.sB2 = 64;
      g.sC0 = (long)48 * 262144; g.sC1 = (long)12 * 262144; g.sC2 = 262144;
      launch_gemm<1>(stream, g, 192); }
    softmax_mask<<<dim3(512, 192), 128, 0, stream>>>(sc);
    // y = P @ V
    { GemmArgs g = ga(sc, 512, qkvb + 1536, 2304, yb, 768, 512, 64, 512);
      g.bn1 = 4; g.bn2 = 12;
      g.sA0 = (long)48 * 262144; g.sA1 = (long)12 * 262144; g.sA2 = 262144;
      g.sB0 = (long)2048 * 2304; g.sB1 = (long)512 * 2304;  g.sB2 = 64;
      g.sC0 = (long)2048 * 768;  g.sC1 = (long)512 * 768;   g.sC2 = 64;
      launch_gemm<0>(stream, g, 192); }
    // per-direction output projection -> concat layout [2048, 4*768]
    { GemmArgs g = ga(yb, 768, aow, 768, catb, 3072, 2048, 768, 768);
      g.sA0 = (long)2048 * 768; g.sB0 = (long)768 * 768; g.sC0 = 768;
      launch_gemm<0>(stream, g, 4); }
    // gate = sigmoid(concat @ gate_w + b)
    { GemmArgs g = ga(catb, 3072, gw, 3072, gateb, 3072, 2048, 3072, 3072);
      g.bias = gb; g.act = 1; launch_gemm<0>(stream, g, 1); }
    // fused = (gate * concat) @ fuse_w   (elementwise A-multiplier fused)
    { GemmArgs g = ga(catb, 3072, fw, 768, fusedb, 768, 2048, 768, 3072);
      g.A2 = gateb; launch_gemm<0>(stream, g, 1); }
    rms_kernel<<<2048, 256, 0, stream>>>(fusedb, h, anw, x1);     // x1 = rms(fused+h)
    rms_kernel<<<2048, 256, 0, stream>>>(x1, nullptr, ln2, hh);
    { GemmArgs g = ga(hh, 768, f1, 3072, ffb, 3072, 2048, 3072, 768);
      g.act = 2; launch_gemm<0>(stream, g, 1); }                   // silu(hh@ff1)
    { GemmArgs g = ga(ffb, 3072, f2, 768, x, 768, 2048, 768, 3072);
      g.res = x1; launch_gemm<0>(stream, g, 1); }                  // x = ff2 + x1
  }

  // ---- cell decoder ----
  { int total = 12288 * 768;
    expand_kernel<<<(total + 255) / 256, 256, 0, stream>>>(x, dec_pos, expb, total); }
  rms_kernel<<<12288, 256, 0, stream>>>(expb, tokres, dn1_w, dech);
  { GemmArgs g = ga(dech, 768, dlin1, 768, dt1, 768, 12288, 768, 768);
    g.act = 2; launch_gemm<0>(stream, g, 1); }
  { GemmArgs g = ga(dt1, 768, dlin2, 768, dtmp, 768, 12288, 768, 768);
    g.res = expb; launch_gemm<0>(stream, g, 1); }
  dec_double_rms<<<12288, 256, 0, stream>>>(dtmp, dn2_w, lnf_w, finalb);
  // ---- tied logits head: final @ tok_emb^T  (tok_emb already [N=V, K=D]) ----
  { GemmArgs g = ga(finalb, 768, tok_emb, 768, out, 256, 12288, 256, 768);
    launch_gemm<1>(stream, g, 1); }
}
